// SAGEConvLayer_76166950027377
// MI455X (gfx1250) — matrix operand order repaired and checked by reading, not compiled, not run on hardware
//
#include <hip/hip_runtime.h>
#include <hip/hip_fp16.h>

typedef __attribute__((ext_vector_type(16))) _Float16 v16h;
typedef __attribute__((ext_vector_type(8)))  float    v8f;

#define DIM       128
#define N_PAPER   100000
#define N_AUTHOR  50000
#define E_CITES   800000
#define E_WRITES  500000
#define E_WB      500000

// ---- workspace layout (in floats) ----
#define OFF_SUM_C   ((size_t)0)
#define OFF_SUM_W   (OFF_SUM_C  + (size_t)N_PAPER  * DIM)   // 12.8M
#define OFF_SUM_WB  (OFF_SUM_W  + (size_t)N_PAPER  * DIM)   // 25.6M
#define OFF_CNT_C   (OFF_SUM_WB + (size_t)N_AUTHOR * DIM)   // 32.0M
#define OFF_CNT_W   (OFF_CNT_C  + (size_t)N_PAPER)
#define OFF_CNT_WB  (OFF_CNT_W  + (size_t)N_PAPER)
#define OFF_BN_PS   (OFF_CNT_WB + (size_t)N_AUTHOR)         // paper sum[128]
#define OFF_BN_PQ   (OFF_BN_PS + 128)                       // paper sumsq[128]
#define OFF_BN_AS   (OFF_BN_PQ + 128)                       // author sum[128]
#define OFF_BN_AQ   (OFF_BN_AS + 128)
#define WS_FLOATS   (OFF_BN_AQ + 128)

// -------------------------------------------------------------------------
// Phase 1: edge scatter. One wave32 per edge: lanes cooperatively read the
// full 512B source row (float4 per lane, contiguous) and atomically add
// into the per-destination sum buffer (L2-resident on MI455X: all hot
// buffers fit in the 192MB L2).
// -------------------------------------------------------------------------
__global__ void scatter_edges(const float* __restrict__ xsrc,
                              const int* __restrict__ src,
                              const int* __restrict__ dst,
                              float* __restrict__ sum,
                              float* __restrict__ cnt,
                              int nedges) {
  int gwave = (int)((blockIdx.x * (size_t)blockDim.x + threadIdx.x) >> 5);
  int lane  = threadIdx.x & 31;
  if (gwave >= nedges) return;
  int s = src[gwave];
  int d = dst[gwave];
  const float4* sp = (const float4*)(xsrc + (size_t)s * DIM);
  float4 v = sp[lane];
  float* dp = sum + (size_t)d * DIM + lane * 4;
  atomicAdd(dp + 0, v.x);
  atomicAdd(dp + 1, v.y);
  atomicAdd(dp + 2, v.z);
  atomicAdd(dp + 3, v.w);
  if (lane == 0) atomicAdd(cnt + d, 1.0f);
}

// -------------------------------------------------------------------------
// WMMA fragment loaders (f32 memory -> f16 registers, contiguous k = 16*half + s, same for A and B).
// A 16x32 f16:  lane = M (0..15 twice); half-wave selects K 0..15 vs
//               16..31, sequential.  p points at row + ko + half*16.
// -------------------------------------------------------------------------
__device__ __forceinline__ v16h load_afrag(const float* __restrict__ p, float scale) {
  v16h f;
#pragma unroll
  for (int i = 0; i < 8; ++i) {
    f[i]     = (_Float16)(p[i]      * scale);
    f[i + 8] = (_Float16)(p[i + 8] * scale);
  }
  return f;
}

// B 32x16 f16: lane = N; half-wave selects K 0..15 vs 16..31, sequential.
// p points at Wrow + ko + half*16; optional second matrix summed in (for
// fusing Wr_cites + Wr_writes on the root path).
__device__ __forceinline__ v16h load_bfrag(const float* __restrict__ p,
                                           const float* __restrict__ q) {
  v16h f;
#pragma unroll
  for (int i = 0; i < 16; ++i) {
    float v = p[i];
    if (q) v += q[i];
    f[i] = (_Float16)v;
  }
  return f;
}

// -------------------------------------------------------------------------
// Phase 2: fused SAGE GEMM.  out[i,j] = sum_k meanS(i,k)*WlS(j,k) over the
// mean sources S, plus x(i,k)*(Wr0+Wr1)(j,k), plus bias.  One workgroup
// (8 waves) computes a 16x128 output slab; each wave owns one 16x16 tile
// and runs K=128 in 4 steps of v_wmma_f32_16x16x32_f16 per source.
// -------------------------------------------------------------------------
__global__ void __launch_bounds__(256)
sage_gemm(const float* __restrict__ A0, const float* __restrict__ cnt0,
          const float* __restrict__ W0,
          const float* __restrict__ A1, const float* __restrict__ cnt1,
          const float* __restrict__ W1,
          const float* __restrict__ X,
          const float* __restrict__ Wr0, const float* __restrict__ Wr1,
          const float* __restrict__ b0, const float* __restrict__ b1,
          float* __restrict__ out, int nrows) {
  const int lane = threadIdx.x & 31;
  const int w    = threadIdx.x >> 5;          // 0..7  -> 16-col tile
  const int half = lane >> 4;
  const int l15  = lane & 15;
  const int R    = blockIdx.x * 16;           // row tile base
  const int rowA = R + l15;                   // A-source row for this lane
  const int colB = w * 16 + l15;              // output column / weight row
  if (R >= nrows) return;

  const float inv0 = 1.0f / fmaxf(cnt0[rowA], 1.0f);
  const float inv1 = A1 ? (1.0f / fmaxf(cnt1[rowA], 1.0f)) : 0.0f;

  const float* a0base = A0 + (size_t)rowA * DIM + half * 16;
  const float* a1base = A1 ? (A1 + (size_t)rowA * DIM + half * 16) : nullptr;
  const float* xbase  = X  + (size_t)rowA * DIM + half * 16;
  const float* w0base = W0 + (size_t)colB * DIM + half * 16;
  const float* w1base = W1 ? (W1 + (size_t)colB * DIM + half * 16) : nullptr;
  const float* r0base = Wr0 + (size_t)colB * DIM + half * 16;
  const float* r1base = Wr1 ? (Wr1 + (size_t)colB * DIM + half * 16) : nullptr;

  v8f c = {};
#pragma unroll
  for (int ko = 0; ko < DIM; ko += 32) {
    // mean source 0
    v16h a = load_afrag(a0base + ko, inv0);
    v16h b = load_bfrag(w0base + ko, nullptr);
    c = __builtin_amdgcn_wmma_f32_16x16x32_f16(false, a, false, b,
                                               (short)0, c, false, false);
    // mean source 1 (paper dst only)
    if (A1) {
      v16h a1 = load_afrag(a1base + ko, inv1);
      v16h b1 = load_bfrag(w1base + ko, nullptr);
      c = __builtin_amdgcn_wmma_f32_16x16x32_f16(false, a1, false, b1,
                                                 (short)0, c, false, false);
    }
    // root path: X @ (Wr0 [+ Wr1])^T
    v16h ax = load_afrag(xbase + ko, 1.0f);
    v16h br = load_bfrag(r0base + ko, r1base ? (r1base + ko) : nullptr);
    c = __builtin_amdgcn_wmma_f32_16x16x32_f16(false, ax, false, br,
                                               (short)0, c, false, false);
  }

  float bias = b0[colB] + (b1 ? b1[colB] : 0.0f);
  // D layout: VGPR v holds row M = half*8 + v, column N = l15
#pragma unroll
  for (int v = 0; v < 8; ++v) {
    out[(size_t)(R + half * 8 + v) * DIM + colB] = c[v] + bias;
  }
}

// -------------------------------------------------------------------------
// Phase 3: BatchNorm stats over ReLU(h).  128 threads/block, one column per
// thread (coalesced row reads); block partials -> f32 atomics.
// -------------------------------------------------------------------------
__global__ void bn_reduce(const float* __restrict__ h, int nrows,
                          float* __restrict__ bnsum, float* __restrict__ bnsq) {
  int col = threadIdx.x;  // blockDim.x == 128
  int chunk = (nrows + gridDim.x - 1) / gridDim.x;
  int r0 = blockIdx.x * chunk;
  int r1 = r0 + chunk; if (r1 > nrows) r1 = nrows;
  float s = 0.0f, q = 0.0f;
  for (int r = r0; r < r1; ++r) {
    float v = fmaxf(h[(size_t)r * DIM + col], 0.0f);
    s += v;
    q += v * v;
  }
  atomicAdd(&bnsum[col], s);
  atomicAdd(&bnsq[col], q);
}

__global__ void bn_apply(float* __restrict__ h, int nrows,
                         const float* __restrict__ bnsum,
                         const float* __restrict__ bnsq,
                         const float* __restrict__ gamma,
                         const float* __restrict__ beta) {
  size_t total  = (size_t)nrows * DIM;
  size_t stride = (size_t)gridDim.x * blockDim.x;
  float invN = 1.0f / (float)nrows;
  for (size_t idx = blockIdx.x * (size_t)blockDim.x + threadIdx.x;
       idx < total; idx += stride) {
    int col = (int)(idx & (DIM - 1));
    float mu  = bnsum[col] * invN;
    float var = bnsq[col] * invN - mu * mu;
    float rs  = rsqrtf(var + 1e-5f);
    float v   = fmaxf(h[idx], 0.0f);
    h[idx] = gamma[col] * (v - mu) * rs + beta[col];
  }
}

// -------------------------------------------------------------------------
extern "C" void kernel_launch(void* const* d_in, const int* in_sizes, int n_in,
                              void* d_out, int out_size, void* d_ws, size_t ws_size,
                              hipStream_t stream) {
  const float* x_paper   = (const float*)d_in[0];
  const float* x_author  = (const float*)d_in[1];
  const int*   cites_src = (const int*)d_in[2];
  const int*   cites_dst = (const int*)d_in[3];
  const int*   writes_src= (const int*)d_in[4];
  const int*   writes_dst= (const int*)d_in[5];
  const int*   wb_src    = (const int*)d_in[6];
  const int*   wb_dst    = (const int*)d_in[7];
  const float* Wl_cites  = (const float*)d_in[8];
  const float* bl_cites  = (const float*)d_in[9];
  const float* Wr_cites  = (const float*)d_in[10];
  const float* Wl_writes = (const float*)d_in[11];
  const float* bl_writes = (const float*)d_in[12];
  const float* Wr_writes = (const float*)d_in[13];
  const float* Wl_wb     = (const float*)d_in[14];
  const float* bl_wb     = (const float*)d_in[15];
  const float* Wr_wb     = (const float*)d_in[16];
  const float* gamma_p   = (const float*)d_in[17];
  const float* beta_p    = (const float*)d_in[18];
  const float* gamma_a   = (const float*)d_in[19];
  const float* beta_a    = (const float*)d_in[20];

  float* ws = (float*)d_ws;
  float* sum_c  = ws + OFF_SUM_C;
  float* sum_w  = ws + OFF_SUM_W;
  float* sum_wb = ws + OFF_SUM_WB;
  float* cnt_c  = ws + OFF_CNT_C;
  float* cnt_w  = ws + OFF_CNT_W;
  float* cnt_wb = ws + OFF_CNT_WB;
  float* bn_ps  = ws + OFF_BN_PS;
  float* bn_pq  = ws + OFF_BN_PQ;
  float* bn_as  = ws + OFF_BN_AS;
  float* bn_aq  = ws + OFF_BN_AQ;

  float* out_paper  = (float*)d_out;
  float* out_author = out_paper + (size_t)N_PAPER * DIM;

  // zero accumulators (graph-capturable memset node)
  hipMemsetAsync(d_ws, 0, WS_FLOATS * sizeof(float), stream);

  // Phase 1: scatter-sum per relation (8 edges/256-thread block)
  scatter_edges<<<(E_CITES + 7) / 8, 256, 0, stream>>>(
      x_paper, cites_src, cites_dst, sum_c, cnt_c, E_CITES);
  scatter_edges<<<(E_WRITES + 7) / 8, 256, 0, stream>>>(
      x_author, writes_src, writes_dst, sum_w, cnt_w, E_WRITES);
  scatter_edges<<<(E_WB + 7) / 8, 256, 0, stream>>>(
      x_paper, wb_src, wb_dst, sum_wb, cnt_wb, E_WB);

  // Phase 2: fused SAGE GEMMs (16 rows per workgroup; counts divide evenly)
  sage_gemm<<<N_PAPER / 16, 256, 0, stream>>>(
      sum_c, cnt_c, Wl_cites,
      sum_w, cnt_w, Wl_writes,
      x_paper, Wr_cites, Wr_writes,
      bl_cites, bl_writes,
      out_paper, N_PAPER);
  sage_gemm<<<N_AUTHOR / 16, 256, 0, stream>>>(
      sum_wb, cnt_wb, Wl_wb,
      nullptr, nullptr, nullptr,
      x_author, Wr_wb, nullptr,
      bl_wb, nullptr,
      out_author, N_AUTHOR);

  // Phase 3: BatchNorm (ReLU folded in)
  bn_reduce<<<512, 128, 0, stream>>>(out_paper, N_PAPER, bn_ps, bn_pq);
  bn_reduce<<<512, 128, 0, stream>>>(out_author, N_AUTHOR, bn_as, bn_aq);
  bn_apply<<<8192, 256, 0, stream>>>(out_paper, N_PAPER, bn_ps, bn_pq, gamma_p, beta_p);
  bn_apply<<<4096, 256, 0, stream>>>(out_author, N_AUTHOR, bn_as, bn_aq, gamma_a, beta_a);
}
